// ShufflerAttention_25228637897008
// MI455X (gfx1250) — matrix operand order repaired and checked by reading, not compiled, not run on hardware
//
#include <hip/hip_runtime.h>
#include <hip/hip_bf16.h>

typedef __attribute__((ext_vector_type(16))) __bf16 v16bf;
typedef __attribute__((ext_vector_type(8)))  __bf16 v8bf;
typedef __attribute__((ext_vector_type(4)))  __bf16 v4bf;
typedef __attribute__((ext_vector_type(8)))  float  v8f;

union V16 { v16bf v; v8bf h[2]; };

constexpr int kB = 4, kNB = 64, kT = 512, kC = 256, kH = 8, kHD = 32;
constexpr int kFW = 8, kTW = 8, kL = 64;
constexpr int kNWN = kNB / kFW;        // 8
constexpr int kNWT = kT / kTW;         // 64
constexpr int kWinPerB = kNWN * kNWT;  // 512
constexpr int kNWin = kB * kWinPerB;   // 2048
constexpr float kScale = 0.17677669529663689f;  // HD^-0.5
constexpr float kEps = 1e-5f;
constexpr int kWElems = kC * kC;       // 65536 bf16 elems per swizzled weight

__device__ __forceinline__ v8f wmma_bf16(v16bf a, v16bf b, v8f c) {
  // D = A(16x32) * B(32x16) + C, fp32 accumulate
  return __builtin_amdgcn_wmma_f32_16x16x32_bf16(false, a, false, b, (short)0, c,
                                                 false, false);
}

// A-fragment (16x32 bf16) from row-major LDS buffer.
// Lane holds row m0+(lane&15); lanes<16 hold K = k0+[0..15],
// lanes>=16 hold K = k0+[16..31].
__device__ __forceinline__ v16bf load_a_frag(const __bf16* buf, int ld,
                                             int m0, int k0, int lane) {
  const int row = m0 + (lane & 15);
  const int kb  = k0 + (lane >> 4) * 16;
  V16 r;
  r.h[0] = *(const v8bf*)(buf + row * ld + kb);
  r.h[1] = *(const v8bf*)(buf + row * ld + kb + 8);
  return r.v;
}

// B-fragment (32x16) where B[k][n] = buf[n][kbase+k] (K contiguous in memory):
// lane holds column n0+(lane&15); lanes<16 hold K=0..15, lanes>=16 hold K=16..31.
__device__ __forceinline__ v16bf load_b_kcontig(const __bf16* buf, int ld,
                                                int n0, int kbase, int lane) {
  const int row = n0 + (lane & 15);
  const int kb  = kbase + (lane >> 4) * 16;
  V16 r;
  r.h[0] = *(const v8bf*)(buf + row * ld + kb);
  r.h[1] = *(const v8bf*)(buf + row * ld + kb + 8);
  return r.v;
}

// B-fragment from pre-swizzled storage: 32 bytes per lane, fully coalesced.
__device__ __forceinline__ v16bf load_sw_frag(const __bf16* p, int lane) {
  V16 r;
  r.h[0] = *(const v8bf*)(p + lane * 16);
  r.h[1] = *(const v8bf*)(p + lane * 16 + 8);
  return r.v;
}

// ---------------------------------------------------------------------------
// Kernel 0: transpose + B-fragment-swizzle the 4 weight matrices into bf16.
// ---------------------------------------------------------------------------
__global__ void prep_weights(const float* __restrict__ Wq, const float* __restrict__ Wk,
                             const float* __restrict__ Wv, const float* __restrict__ Wp,
                             __bf16* __restrict__ wsw) {
  const int idx = blockIdx.x * blockDim.x + threadIdx.x;
  const int wgt = idx >> 16;
  const int rem = idx & 0xFFFF;
  const int i    = rem & 15;
  const int lane = (rem >> 4) & 31;
  const int nt   = (rem >> 9) & 15;
  const int ks   = rem >> 13;
  const int k = ks * 32 + (lane >> 4) * 16 + i;
  const int n = nt * 16 + (lane & 15);
  const float* W = (wgt == 0) ? Wq : (wgt == 1) ? Wk : (wgt == 2) ? Wv : Wp;
  float v = W[n * kC + k];
  if (wgt == 0) v *= kScale;  // fold softmax scale into Wq (and bq at use site)
  wsw[idx] = (__bf16)v;
}

// ---------------------------------------------------------------------------
// Kernel 1: fully fused shuffler attention. One block = one 64-token window.
// ---------------------------------------------------------------------------
__global__ void __launch_bounds__(256)
shuffler_attention(const float* __restrict__ inpt,
                   const int* __restrict__ perm_n, const int* __restrict__ perm_t,
                   const float* __restrict__ ln_g, const float* __restrict__ ln_b,
                   const float* __restrict__ bq, const float* __restrict__ bk,
                   const float* __restrict__ bv, const float* __restrict__ bp,
                   const __bf16* __restrict__ wsw,
                   float* __restrict__ out) {
  // 224 KB LDS (fits CDNA5's 320 KB WGP LDS)
  __shared__ __align__(32) __bf16 Xn[kL * kC];        // 32 KB LN'd activations
  __shared__ __align__(32) __bf16 Qs[kL * kC];        // 32 KB Q (scaled)
  __shared__ __align__(32) __bf16 Ks[kL * kC];        // 32 KB K
  __shared__ __align__(32) __bf16 Vsw[kH * 2048];     // 32 KB V (B-swizzled)
  __shared__ __align__(32) __bf16 Pl[kH * kL * kL];   // 64 KB probs / fp32 stage
  __shared__ __align__(32) __bf16 Ol[kL * kC];        // 32 KB attn output

  const int tid  = threadIdx.x;
  const int wave = tid >> 5;
  const int lane = tid & 31;
  const int hf   = lane >> 4;
  const int ln16 = lane & 15;

  const int w   = blockIdx.x;
  const int bat = w / kWinPerB;
  const int wn  = (w / kNWT) % kNWN;
  const int wt  = w % kNWT;

  // ---- Phase 1: async gather to LDS + LayerNorm -> Xn (bf16) ----
  {
    float* Xf = (float*)(void*)Pl;  // 64 KB raw fp32 stage, aliased over Pl
    // Wave `wave` async-copies rows wave*8 .. wave*8+7 (16 half-row transfers,
    // 16 B per lane) straight into LDS via the CDNA5 async-DMA path.
    for (int it = 0; it < 16; ++it) {
      const int r  = wave * 8 + (it >> 1);
      const int hh = it & 1;
      const int n_src = perm_n[bat * kNB + wn * kFW + (r >> 3)];
      const int t_src = perm_t[bat * kT + wt * kTW + (r & 7)];
      const float* gsrc = inpt +
          (((size_t)bat * kNB + n_src) * kT + t_src) * kC + hh * 128 + lane * 4;
      const unsigned lds_off =
          (unsigned)(uintptr_t)(Xf + r * kC + hh * 128 + lane * 4);
      const unsigned long long ga = (unsigned long long)(uintptr_t)gsrc;
      asm volatile("global_load_async_to_lds_b128 %0, %1, off"
                   :: "v"(lds_off), "v"(ga) : "memory");
    }
    asm volatile("s_wait_asynccnt 0x0" ::: "memory");
    // LayerNorm out of LDS: 4 threads per row, interleaved float4.
    const int r  = tid >> 2;
    const int qq = tid & 3;
    const float4* rowf = (const float4*)(Xf + r * kC);
    float4 vals[16];
    float s = 0.f, ss = 0.f;
#pragma unroll
    for (int j = 0; j < 16; ++j) {
      float4 v = rowf[qq + j * 4];
      vals[j] = v;
      s  += v.x + v.y + v.z + v.w;
      ss += v.x * v.x + v.y * v.y + v.z * v.z + v.w * v.w;
    }
    s  += __shfl_xor(s, 1);  s  += __shfl_xor(s, 2);
    ss += __shfl_xor(ss, 1); ss += __shfl_xor(ss, 2);
    const float mean = s * (1.f / kC);
    const float rstd = rsqrtf(ss * (1.f / kC) - mean * mean + kEps);
#pragma unroll
    for (int j = 0; j < 16; ++j) {
      const int c4 = qq + j * 4;
      const float4 v  = vals[j];
      const float4 g  = ((const float4*)ln_g)[c4];
      const float4 be = ((const float4*)ln_b)[c4];
      v4bf o;
      o[0] = (__bf16)((v.x - mean) * rstd * g.x + be.x);
      o[1] = (__bf16)((v.y - mean) * rstd * g.y + be.y);
      o[2] = (__bf16)((v.z - mean) * rstd * g.z + be.z);
      o[3] = (__bf16)((v.w - mean) * rstd * g.w + be.w);
      *(v4bf*)&Xn[r * kC + c4 * 4] = o;
    }
  }
  __syncthreads();

  // ---- Phase 2: Q/K/V = Xn @ W^T + b ----
  // k-step OUTER: each weight B-fragment is loaded from L2 exactly once per
  // block and reused across all 4 M-tiles (24 live accumulators, ~192 VGPRs;
  // fine at 2 waves/SIMD). Weight traffic: 384 KB/block instead of 1.5 MB.
  {
    float biasv[3][2];
#pragma unroll
    for (int m = 0; m < 3; ++m) {
      const float* bm = (m == 0) ? bq : (m == 1) ? bk : bv;
#pragma unroll
      for (int ntl = 0; ntl < 2; ++ntl) {
        float bsv = bm[(wave + ntl * 8) * 16 + ln16];
        biasv[m][ntl] = (m == 0) ? bsv * kScale : bsv;
      }
    }
    v8f acc[3][2][4];
#pragma unroll
    for (int m = 0; m < 3; ++m)
#pragma unroll
      for (int ntl = 0; ntl < 2; ++ntl) {
        const float bsv = biasv[m][ntl];
        v8f binit = {bsv, bsv, bsv, bsv, bsv, bsv, bsv, bsv};
#pragma unroll
        for (int mi = 0; mi < 4; ++mi) acc[m][ntl][mi] = binit;
      }
    for (int ks = 0; ks < 8; ++ks) {
      v16bf a[4];
#pragma unroll
      for (int mi = 0; mi < 4; ++mi)
        a[mi] = load_a_frag(Xn, kC, mi * 16, ks * 32, lane);
#pragma unroll
      for (int m = 0; m < 3; ++m)
#pragma unroll
        for (int ntl = 0; ntl < 2; ++ntl) {
          const int nt = wave + ntl * 8;
          const v16bf bfrag =
              load_sw_frag(wsw + m * kWElems + (ks * 16 + nt) * 512, lane);
#pragma unroll
          for (int mi = 0; mi < 4; ++mi)
            acc[m][ntl][mi] = wmma_bf16(a[mi], bfrag, acc[m][ntl][mi]);
        }
    }
#pragma unroll
    for (int m = 0; m < 3; ++m)
#pragma unroll
      for (int ntl = 0; ntl < 2; ++ntl) {
        const int col = (wave + ntl * 8) * 16 + ln16;
#pragma unroll
        for (int mi = 0; mi < 4; ++mi)
#pragma unroll
          for (int j = 0; j < 8; ++j) {
            const int row = mi * 16 + j + hf * 8;
            const __bf16 val = (__bf16)acc[m][ntl][mi][j];
            if (m == 0)      Qs[row * kC + col] = val;
            else if (m == 1) Ks[row * kC + col] = val;
            else {
              // scatter V into B-fragment-swizzled layout (per head)
              const int hh = col >> 5, dl = col & 31;
              const int tk = row >> 5, ii = row & 15, lh = (row >> 4) & 1;
              Vsw[hh * 2048 + (tk * 2 + (dl >> 4)) * 512 +
                  (lh * 16 + (dl & 15)) * 16 + ii] = val;
            }
          }
      }
  }
  __syncthreads();

  // ---- Phase 3: per-head attention (head = wave) ----
  {
    const int h = wave;
    __bf16* Ph = Pl + h * (kL * kL);
    // scores + softmax, one 16-row block at a time (32 VGPRs of scores live)
    for (int mi = 0; mi < 4; ++mi) {
      const v16bf aq = load_a_frag(Qs, kC, mi * 16, h * kHD, lane);
      v8f s[4];
#pragma unroll
      for (int ni = 0; ni < 4; ++ni) {
        v8f z = {0.f, 0.f, 0.f, 0.f, 0.f, 0.f, 0.f, 0.f};
        s[ni] = wmma_bf16(aq, load_b_kcontig(Ks, kC, ni * 16, h * kHD, lane), z);
      }
#pragma unroll
      for (int j = 0; j < 8; ++j) {
        float mx = fmaxf(fmaxf(s[0][j], s[1][j]), fmaxf(s[2][j], s[3][j]));
#pragma unroll
        for (int o = 8; o >= 1; o >>= 1) mx = fmaxf(mx, __shfl_xor(mx, o, 16));
        float e[4], sum = 0.f;
#pragma unroll
        for (int ni = 0; ni < 4; ++ni) { e[ni] = __expf(s[ni][j] - mx); sum += e[ni]; }
#pragma unroll
        for (int o = 8; o >= 1; o >>= 1) sum += __shfl_xor(sum, o, 16);
        const float rs = __builtin_amdgcn_rcpf(sum);
        const int row = mi * 16 + j + hf * 8;
#pragma unroll
        for (int ni = 0; ni < 4; ++ni)
          Ph[row * kL + ni * 16 + ln16] = (__bf16)(e[ni] * rs);
      }
    }
    // O_h = P @ V_h  (M=64, N=32, K=64)
    for (int mi = 0; mi < 4; ++mi) {
#pragma unroll
      for (int nt2 = 0; nt2 < 2; ++nt2) {
        v8f o = {0.f, 0.f, 0.f, 0.f, 0.f, 0.f, 0.f, 0.f};
#pragma unroll
        for (int tk = 0; tk < 2; ++tk) {
          const v16bf ap  = load_a_frag(Ph, kL, mi * 16, tk * 32, lane);
          const v16bf bvf = load_sw_frag(Vsw + h * 2048 + (tk * 2 + nt2) * 512, lane);
          o = wmma_bf16(ap, bvf, o);
        }
#pragma unroll
        for (int j = 0; j < 8; ++j)
          Ol[(mi * 16 + j + hf * 8) * kC + h * kHD + nt2 * 16 + ln16] =
              (__bf16)o[j];
      }
    }
  }
  __syncthreads();

  // ---- Phase 4: Y = Ol @ Wp^T + bp; residual + inverse-perm scatter store ----
  // k-step outer again: each Wp fragment read once per block, reused over mi.
  {
    const __bf16* Wpw = wsw + 3 * kWElems;
    for (int ntl = 0; ntl < 2; ++ntl) {
      const int nt  = wave + ntl * 8;
      const int col = nt * 16 + ln16;
      const float bias = bp[col];
      v8f acc[4];
#pragma unroll
      for (int mi = 0; mi < 4; ++mi) {
        v8f binit = {bias, bias, bias, bias, bias, bias, bias, bias};
        acc[mi] = binit;
      }
      for (int ks = 0; ks < 8; ++ks) {
        const v16bf bfrag = load_sw_frag(Wpw + (ks * 16 + nt) * 512, lane);
#pragma unroll
        for (int mi = 0; mi < 4; ++mi)
          acc[mi] = wmma_bf16(load_a_frag(Ol, kC, mi * 16, ks * 32, lane),
                              bfrag, acc[mi]);
      }
#pragma unroll
      for (int mi = 0; mi < 4; ++mi) {
        // rows l = mi*16 + hf*8 + j  =>  l>>3 == 2*mi + hf (constant over j)
        const int n_src = perm_n[bat * kNB + wn * kFW + (2 * mi + hf)];
        const size_t base = ((size_t)bat * kNB + n_src) * kT;
#pragma unroll
        for (int j = 0; j < 8; ++j) {
          const int t_src = perm_t[bat * kT + wt * kTW + j];
          const size_t addr = (base + t_src) * kC + col;
          out[addr] = inpt[addr] + acc[mi][j];
        }
      }
    }
  }
}

extern "C" void kernel_launch(void* const* d_in, const int* in_sizes, int n_in,
                              void* d_out, int out_size, void* d_ws, size_t ws_size,
                              hipStream_t stream) {
  (void)in_sizes; (void)n_in; (void)out_size; (void)ws_size;
  const float* inpt   = (const float*)d_in[0];
  const int*   perm_n = (const int*)d_in[1];
  const int*   perm_t = (const int*)d_in[2];
  const float* ln_g   = (const float*)d_in[3];
  const float* ln_b   = (const float*)d_in[4];
  const float* Wq     = (const float*)d_in[5];
  const float* bq     = (const float*)d_in[6];
  const float* Wk     = (const float*)d_in[7];
  const float* bk     = (const float*)d_in[8];
  const float* Wv     = (const float*)d_in[9];
  const float* bv     = (const float*)d_in[10];
  const float* Wp     = (const float*)d_in[11];
  const float* bp     = (const float*)d_in[12];
  __bf16* wsw = (__bf16*)d_ws;   // 4 * 128 KB swizzled bf16 weights
  float* out = (float*)d_out;

  prep_weights<<<(4 * kWElems) / 256, 256, 0, stream>>>(Wq, Wk, Wv, Wp, wsw);
  shuffler_attention<<<kNWin, 256, 0, stream>>>(inpt, perm_n, perm_t, ln_g, ln_b,
                                                bq, bk, bv, bp, wsw, out);
}
